// LateralLayerEfficient_6433861009580
// MI455X (gfx1250) — matrix operand order repaired and checked by reading, not compiled, not run on hardware
//
#include <hip/hip_runtime.h>

// ---------------------------------------------------------------------------
// Reference: conv5x5 SAME (B=16,Cin=64,Cout=128,H=W=128) - bias, k-WTA (k=4)
// across channels, per-batch rank-3276 (0-based desc) threshold, binary mask.
// Outputs: realy (33.5M f32) ++ binary (33.5M f32).
// ---------------------------------------------------------------------------
#define BATCH   16
#define CIN     64
#define COUT    128
#define HH      128
#define WW      128
#define KTOT    (CIN * 25)          // 1600
#define KSTEPS  (KTOT / 32)         // 50
#define HW      (HH * WW)           // 16384
#define PIX_PER_BATCH (COUT * HW)   // 2,097,152
#define NOUT    (BATCH * PIX_PER_BATCH)
#define NX      (BATCH * CIN * HW)  // 16,777,216
#define RANK1   3277                // 1-based k-th largest (index 3276)
#define CAP     131072              // compact survivor capacity per batch
#define WGCAP   1024                // survivor capacity per workgroup
#define ZKEY    0x80000000u         // sort key of +0.0f
#define NEGBIG  (-3.402823466e38f)

typedef __attribute__((ext_vector_type(16))) __bf16 v16bf;
typedef __attribute__((ext_vector_type(8)))  float  v8f;

union Frag16 {
  unsigned int u[8];
  v16bf        v;
};

static __device__ __forceinline__ unsigned short f32_to_bf16(float f) {
  unsigned int u = __float_as_uint(f);
  unsigned int lsb = (u >> 16) & 1u;
  u += 0x7fffu + lsb;                       // round-to-nearest-even
  return (unsigned short)(u >> 16);
}

static __device__ __forceinline__ unsigned int f32_key(float f) {
  unsigned int u = __float_as_uint(f);
  return (u & 0x80000000u) ? ~u : (u | 0x80000000u);
}

static __device__ __forceinline__ void top4_insert(float v, float& m1,
                                                   float& m2, float& m3,
                                                   float& m4) {
  if (v > m4) {
    if (v > m1)      { m4 = m3; m3 = m2; m2 = m1; m1 = v; }
    else if (v > m2) { m4 = m3; m3 = m2; m2 = v; }
    else if (v > m3) { m4 = m3; m3 = v; }
    else             { m4 = v; }
  }
}

// ---------------------------------------------------------------------------
// Pre-pack kernels: x -> bf16 (L2-resident working set), W -> bf16 in
// chunk-contiguous [ks][m][k] order for coalesced in-loop staging.
// ---------------------------------------------------------------------------
__global__ void __launch_bounds__(256)
pack_x_kernel(const float* __restrict__ X, unsigned short* __restrict__ Xb) {
  int i = blockIdx.x * 256 + threadIdx.x;      // one float4 per thread
  if (i * 4 >= NX) return;
  float4 f = ((const float4*)X)[i];
  unsigned long long p =
      (unsigned long long)f32_to_bf16(f.x) |
      ((unsigned long long)f32_to_bf16(f.y) << 16) |
      ((unsigned long long)f32_to_bf16(f.z) << 32) |
      ((unsigned long long)f32_to_bf16(f.w) << 48);
  ((unsigned long long*)Xb)[i] = p;
}

__global__ void __launch_bounds__(256)
pack_w_kernel(const float* __restrict__ Wg, unsigned short* __restrict__ Wpk) {
  int i = blockIdx.x * 256 + threadIdx.x;      // dst index [ks][m][k]
  if (i >= COUT * KTOT) return;
  int ks = i >> 12;                            // / 4096
  int r  = i & 4095;
  int m  = r >> 5;
  int k  = r & 31;
  Wpk[i] = f32_to_bf16(Wg[(size_t)m * KTOT + ks * 32 + k]);
}

// ---------------------------------------------------------------------------
// Kernel: conv as implicit GEMM via v_wmma_f32_16x16x32_bf16 + fused k-WTA.
// Grid (WW/64, HH, BATCH), 256 threads (8 waves). Each WG: 128 Cout x 64 px.
// W chunks are register-pipelined (next chunk's global loads in flight while
// current chunk's 4 WMMAs per wave execute).
// ---------------------------------------------------------------------------
__global__ void __launch_bounds__(256)
conv_wmma_fused_kernel(const unsigned short* __restrict__ Xb,
                       const unsigned short* __restrict__ Wpk,
                       const float* __restrict__ bias,
                       float* __restrict__ Y, float* __restrict__ binry,
                       float* __restrict__ kthBuf,
                       unsigned int* __restrict__ keys,
                       unsigned int* __restrict__ idxs,
                       unsigned int* __restrict__ cnt) {
  __attribute__((aligned(16))) __shared__ char smem[56576];
  // phase 1 (GEMM):   slab 21760 u16 | wlds 4352 u16 | blds 2176 u16
  unsigned short* slab = (unsigned short*)smem;
  unsigned short* wlds = (unsigned short*)(smem + 43520);
  unsigned short* blds = (unsigned short*)(smem + 52224);
  // phase 2 (epilogue; trailing K-loop barrier protects the aliasing):
  float*        pix   = (float*)smem;                    // [128][64]
  float*        kthl  = (float*)(smem + 32768);          // [64]
  float*        cand  = (float*)(smem + 33024);          // [64][4][4]
  unsigned int* lkeys = (unsigned int*)(smem + 37120);   // [WGCAP]
  unsigned int* lidx  = (unsigned int*)(smem + 41216);   // [WGCAP]
  unsigned int* lmeta = (unsigned int*)(smem + 45312);   // [2]: count, base

  const int tid  = threadIdx.x;
  const int lane = tid & 31;
  const int wave = tid >> 5;
  const int w0   = blockIdx.x * 64;
  const int h    = blockIdx.y;
  const int b    = blockIdx.z;
  const int mg   = wave >> 1;         // 0..3 -> M strips {2mg, 2mg+1}
  const int ng   = wave & 1;          // 0..1 -> N tiles  {2ng, 2ng+1}

  // ---- stage zero-padded bf16 x slab (Cin x 5 x 68) ----
  const int SLAB = CIN * 5 * 68;
  for (int i = tid; i < SLAB; i += 256) {
    int ci = i / (5 * 68);
    int rr = (i / 68) % 5;
    int cc = i % 68;
    int gh = h + rr - 2;
    int gw = w0 + cc - 2;
    unsigned short v = 0;
    if (gh >= 0 && gh < HH && gw >= 0 && gw < WW)
      v = Xb[(((size_t)b * CIN + ci) * HH + gh) * WW + gw];
    slab[i] = v;
  }

  v8f acc[2][2];
  #pragma unroll
  for (int aa = 0; aa < 2; ++aa)
    #pragma unroll
    for (int cc = 0; cc < 2; ++cc) acc[aa][cc] = (v8f){0,0,0,0,0,0,0,0};

  const int halfK8  = (lane >> 4) << 3;   // A frag K-half select
  const int halfK16 = (lane >> 4) << 4;   // unused (B frag selects with halfK8)
  const int lm      = lane & 15;

  // register pipeline for W chunks (each chunk = 2048 u32 = 8 per thread)
  const unsigned int* Wp32 = (const unsigned int*)Wpk;
  unsigned int wreg[8];
  #pragma unroll
  for (int j = 0; j < 8; ++j) wreg[j] = Wp32[tid + j * 256];

  for (int ks = 0; ks < KSTEPS; ++ks) {
    const int k0 = ks * 32;
    // stage current W chunk from registers into padded LDS layout
    #pragma unroll
    for (int j = 0; j < 8; ++j) {
      int i = tid + j * 256;              // 0..2047
      int m = i >> 4, kp = i & 15;
      *(unsigned int*)&wlds[m * 34 + kp * 2] = wreg[j];
    }
    // im2col chunk: 64 px x 32 K gathered from slab
    for (int i = tid; i < 64 * 32; i += 256) {
      int n = i >> 5, k = i & 31;
      int kg = k0 + k;
      int ci = kg / 25;
      int rem = kg - ci * 25;
      int kh = rem / 5;
      int kw = rem - kh * 5;
      blds[n * 34 + k] = slab[ci * 340 + kh * 68 + n + kw];
    }
    // issue next W chunk's global loads (latency hidden under WMMAs)
    if (ks + 1 < KSTEPS) {
      #pragma unroll
      for (int j = 0; j < 8; ++j)
        wreg[j] = Wp32[(ks + 1) * 2048 + tid + j * 256];
    }
    __syncthreads();

    // A fragments (ISA 16-bit 16x32 layout): lanes 0-15 M=lane K{0-7,16-23};
    // lanes 16-31 M=lane-16 K{8-15,24-31}; 2 bf16 per VGPR.
    Frag16 a[2];
    #pragma unroll
    for (int aa = 0; aa < 2; ++aa) {
      const unsigned short* wrow = &wlds[((mg * 2 + aa) * 16 + lm) * 34];
      #pragma unroll
      for (int j = 0; j < 4; ++j)
        a[aa].u[j] = *(const unsigned int*)&wrow[halfK8 + 2 * j];
      #pragma unroll
      for (int j = 0; j < 4; ++j)
        a[aa].u[4 + j] = *(const unsigned int*)&wrow[halfK8 + 16 + 2 * j];
    }
    #pragma unroll
    for (int cc = 0; cc < 2; ++cc) {
      // B fragment (32x16): lanes 0-15 K{0-7,16-23} (col=lane), lanes 16-31 K{8-15,24-31}
      Frag16 bf;
      const unsigned short* brow =
          &blds[((ng * 2 + cc) * 16 + lm) * 34 + halfK8];
      #pragma unroll
      for (int j = 0; j < 8; ++j)
        bf.u[j] = *(const unsigned int*)&brow[2 * j + (j >= 4 ? 8 : 0)];
      #pragma unroll
      for (int aa = 0; aa < 2; ++aa)
        acc[aa][cc] = __builtin_amdgcn_wmma_f32_16x16x32_bf16(
            false, a[aa].v, false, bf.v, (short)0, acc[aa][cc], false, false);
    }
    __syncthreads();
  }

  // ---- epilogue: realy + binary-zero stores, stage pixels to LDS ----
  const int mh = (lane >> 4) << 3;       // C frag: lanes 16-31 hold M = 8+r
  #pragma unroll
  for (int aa = 0; aa < 2; ++aa) {
    #pragma unroll
    for (int cc = 0; cc < 2; ++cc) {
      int n  = (ng * 2 + cc) * 16 + lm;
      int wp = w0 + n;
      #pragma unroll
      for (int r = 0; r < 8; ++r) {
        int m = (mg * 2 + aa) * 16 + mh + r;
        float val = acc[aa][cc][r] - bias[m];
        size_t gi = (((size_t)b * COUT + m) * HH + h) * WW + wp;
        Y[gi]     = val;
        binry[gi] = 0.0f;                 // scatter pass sets the ones
        pix[m * 64 + n] = val;
      }
    }
  }
  if (tid == 0) lmeta[0] = 0;
  __syncthreads();

  // per-pixel top-4: 4 threads/pixel partials, then 64-thread merge
  {
    int n = tid & 63, part = tid >> 6;
    float m1 = NEGBIG, m2 = NEGBIG, m3 = NEGBIG, m4 = NEGBIG;
    for (int c = part * 32; c < part * 32 + 32; ++c)
      top4_insert(pix[c * 64 + n], m1, m2, m3, m4);
    float* cd = &cand[(n * 4 + part) * 4];
    cd[0] = m1; cd[1] = m2; cd[2] = m3; cd[3] = m4;
  }
  __syncthreads();
  if (tid < 64) {
    float m1 = NEGBIG, m2 = NEGBIG, m3 = NEGBIG, m4 = NEGBIG;
    for (int j = 0; j < 16; ++j)
      top4_insert(cand[tid * 16 + j], m1, m2, m3, m4);
    kthl[tid] = m4;
    kthBuf[b * HW + h * WW + w0 + tid] = m4;
  }
  __syncthreads();

  // collect survivors (v >= kth) into LDS, then flush compactly
  for (int i = tid; i < COUT * 64; i += 256) {
    int m = i >> 6, n = i & 63;
    float v = pix[i];
    if (v >= kthl[n]) {
      unsigned int s = atomicAdd(&lmeta[0], 1u);
      if (s < WGCAP) {
        lkeys[s] = f32_key(v);
        lidx[s]  = (unsigned int)(m * HW + h * WW + w0 + n);
      }
    }
  }
  __syncthreads();
  if (tid == 0) {
    unsigned int t = lmeta[0];
    if (t > WGCAP) t = WGCAP;
    lmeta[0] = t;
    lmeta[1] = atomicAdd(&cnt[b], t);    // reserve global range
  }
  __syncthreads();
  unsigned int tot  = lmeta[0];
  unsigned int base = lmeta[1];
  for (unsigned int i = tid; i < tot; i += 256) {
    unsigned int g = base + i;
    if (g < CAP) {
      keys[(size_t)b * CAP + g] = lkeys[i];
      idxs[(size_t)b * CAP + g] = lidx[i];
    }
  }
}

// ---------------------------------------------------------------------------
// Selection init: zero hist & counters, rank state.
// ---------------------------------------------------------------------------
__global__ void sel_init_kernel(unsigned int* hist, unsigned int* prefix,
                                unsigned int* want, unsigned int* cnt) {
  int t = blockIdx.x * 256 + threadIdx.x;
  if (t < BATCH * 256) hist[t] = 0;
  if (t < BATCH) { prefix[t] = 0; want[t] = RANK1; cnt[t] = 0; }
}

// ---------------------------------------------------------------------------
// Radix-select round p (byte MSB->LSB) over compact survivor keys; the ~2M
// masked zeros per batch are folded in analytically in the scan kernel.
// ---------------------------------------------------------------------------
__global__ void __launch_bounds__(256)
sel_hist_kernel(const unsigned int* __restrict__ keys,
                const unsigned int* __restrict__ cnt,
                unsigned int* __restrict__ hist,
                const unsigned int* __restrict__ prefix, int p) {
  __shared__ unsigned int lh[256];
  lh[threadIdx.x] = 0;
  __syncthreads();
  const int b = blockIdx.y;
  unsigned int n = cnt[b];
  if (n > CAP) n = CAP;
  const unsigned int pref = prefix[b];
  const unsigned int* kb = keys + (size_t)b * CAP;
  const unsigned int stride = gridDim.x * 256;
  for (unsigned int i = blockIdx.x * 256 + threadIdx.x; i < n; i += stride) {
    unsigned int key = kb[i];
    bool match = true;
    if (p < 3) {
      int sh = (p + 1) * 8;
      match = ((key >> sh) == (pref >> sh));
    }
    if (match) atomicAdd(&lh[(key >> (p * 8)) & 255u], 1u);
  }
  __syncthreads();
  if (lh[threadIdx.x]) atomicAdd(&hist[b * 256 + threadIdx.x], lh[threadIdx.x]);
}

__global__ void sel_scan_kernel(unsigned int* hist, unsigned int* prefix,
                                unsigned int* want, unsigned int* thrKey,
                                const unsigned int* cnt, int p) {
  if (threadIdx.x < BATCH) {
    const int b = threadIdx.x;
    unsigned int w    = want[b];
    unsigned int pref = prefix[b];
    unsigned int zc   = (unsigned int)PIX_PER_BATCH - cnt[b];  // masked zeros
    bool zmatch = true;
    if (p < 3) {
      int sh = (p + 1) * 8;
      zmatch = ((ZKEY >> sh) == (pref >> sh));
    }
    const int binz = (int)((ZKEY >> (p * 8)) & 255u);
    const unsigned int* hb = &hist[b * 256];
    int bin = 255;
    for (; bin > 0; --bin) {
      unsigned int c = hb[bin];
      if (zmatch && bin == binz) c += zc;
      if (c >= w) break;
      w -= c;
    }
    want[b]   = w;
    pref     |= ((unsigned int)bin << (p * 8));
    prefix[b] = pref;
    if (p == 0) thrKey[b] = pref;
  }
  __syncthreads();
  for (int i = threadIdx.x; i < BATCH * 256; i += blockDim.x) hist[i] = 0;
}

// ---------------------------------------------------------------------------
// Scatter: binary = 1 at surviving indices with key > thr (strict).
// ---------------------------------------------------------------------------
__global__ void __launch_bounds__(256)
scatter_kernel(const unsigned int* __restrict__ keys,
               const unsigned int* __restrict__ idxs,
               const unsigned int* __restrict__ cnt,
               const unsigned int* __restrict__ thrKey,
               float* __restrict__ binry) {
  const int b = blockIdx.y;
  unsigned int n = cnt[b];
  if (n > CAP) n = CAP;
  const unsigned int tk = thrKey[b];
  const unsigned int stride = gridDim.x * 256;
  for (unsigned int i = blockIdx.x * 256 + threadIdx.x; i < n; i += stride) {
    if (keys[(size_t)b * CAP + i] > tk)
      binry[(size_t)b * PIX_PER_BATCH + idxs[(size_t)b * CAP + i]] = 1.0f;
  }
}

// ---------------------------------------------------------------------------
// Fixup (rare): if thr < 0.0, masked zeros satisfy masked > thr; recompute the
// whole batch. Blocks early-exit otherwise (near-zero cost).
// ---------------------------------------------------------------------------
__global__ void __launch_bounds__(256)
fixup_kernel(const float* __restrict__ Y, const float* __restrict__ kthBuf,
             const unsigned int* __restrict__ thrKey,
             float* __restrict__ binry) {
  const int b = blockIdx.y;
  const unsigned int tk = thrKey[b];
  if (ZKEY <= tk) return;              // common case: zeros stay 0
  int i = blockIdx.x * 256 + threadIdx.x;
  if (i >= PIX_PER_BATCH) return;
  int hw = i & (HW - 1);
  float v = Y[(size_t)b * PIX_PER_BATCH + i];
  float m = (v < kthBuf[b * HW + hw]) ? 0.0f : v;
  binry[(size_t)b * PIX_PER_BATCH + i] = (f32_key(m) > tk) ? 1.0f : 0.0f;
}

// ---------------------------------------------------------------------------
extern "C" void kernel_launch(void* const* d_in, const int* in_sizes, int n_in,
                              void* d_out, int out_size, void* d_ws,
                              size_t ws_size, hipStream_t stream) {
  const float* X    = (const float*)d_in[0];   // (16,64,128,128)
  const float* Wg   = (const float*)d_in[1];   // (128,64,5,5)
  const float* bias = (const float*)d_in[2];   // (1,128,1,1)
  // d_in[3] = k (== 4 per setup; top-4 path hardcoded)

  float* realy = (float*)d_out;                // internal activations
  float* binry = (float*)d_out + NOUT;         // binary mask

  // workspace layout (~52 MB)
  char* ws = (char*)d_ws;
  unsigned short* Xb  = (unsigned short*)ws;                  // NX bf16
  unsigned short* Wpk = (unsigned short*)(ws + (size_t)NX * 2);
  char* ws2 = ws + (size_t)NX * 2 + (size_t)COUT * KTOT * 2;
  unsigned int* keys   = (unsigned int*)ws2;                  // 16*CAP
  unsigned int* idxs   = keys + (size_t)BATCH * CAP;          // 16*CAP
  float*        kthBuf = (float*)(idxs + (size_t)BATCH * CAP);
  unsigned int* hist   = (unsigned int*)(kthBuf + BATCH * HW);
  unsigned int* prefix = hist + BATCH * 256;
  unsigned int* want   = prefix + BATCH;
  unsigned int* thrKey = want + BATCH;
  unsigned int* cnt    = thrKey + BATCH;

  pack_x_kernel<<<NX / 4 / 256, 256, 0, stream>>>(X, Xb);
  pack_w_kernel<<<(COUT * KTOT + 255) / 256, 256, 0, stream>>>(Wg, Wpk);
  sel_init_kernel<<<BATCH, 256, 0, stream>>>(hist, prefix, want, cnt);

  conv_wmma_fused_kernel<<<dim3(WW / 64, HH, BATCH), 256, 0, stream>>>(
      Xb, Wpk, bias, realy, binry, kthBuf, keys, idxs, cnt);

  for (int p = 3; p >= 0; --p) {
    sel_hist_kernel<<<dim3(8, BATCH), 256, 0, stream>>>(keys, cnt, hist,
                                                        prefix, p);
    sel_scan_kernel<<<1, 1024, 0, stream>>>(hist, prefix, want, thrKey, cnt, p);
  }

  scatter_kernel<<<dim3(64, BATCH), 256, 0, stream>>>(keys, idxs, cnt, thrKey,
                                                      binry);
  fixup_kernel<<<dim3((PIX_PER_BATCH + 255) / 256, BATCH), 256, 0, stream>>>(
      realy, kthBuf, thrKey, binry);
}
